// TransformerBlock_32478542693203
// MI455X (gfx1250) — matrix operand order repaired and checked by reading, not compiled, not run on hardware
//
#include <hip/hip_runtime.h>
#include <cstdint>
#include <cstddef>

// ---------------------------------------------------------------------------
// Types for CDNA5 WMMA (gfx1250, wave32)
// ---------------------------------------------------------------------------
typedef __attribute__((ext_vector_type(16))) __bf16 bf16x16;
typedef __attribute__((ext_vector_type(8)))  __bf16 bf16x8;
typedef __attribute__((ext_vector_type(8)))  float  floatx8;

#define T_SEQ   2048
#define DIM     2560
#define N_HEADS 20
#define N_KV    5
#define HEAD_D  128
#define QKV_D   3840      // (20 + 2*5) * 128
#define FFN     6912

__device__ __forceinline__ unsigned short f2bf(float f) {
    union { float f; unsigned int u; } x; x.f = f;
    unsigned int r = x.u + 0x7FFFu + ((x.u >> 16) & 1u);   // round-to-nearest-even
    return (unsigned short)(r >> 16);
}

__device__ __forceinline__ bf16x16 pack16(bf16x8 lo, bf16x8 hi) {
    bf16x16 r;
#pragma unroll
    for (int i = 0; i < 8; ++i) { r[i] = lo[i]; r[i + 8] = hi[i]; }
    return r;
}

// ---------------------------------------------------------------------------
// CDNA5 async global->LDS DMA (no VGPR staging, tracked by ASYNCcnt).
// GVS addressing: mem = SGPR64 base + VGPR int32 byte offset + inst offset.
// The LDS destination address is the low 32 bits of the generic pointer
// (flat->LDS mapping truncates to addr[31:0]).
// ---------------------------------------------------------------------------
__device__ __forceinline__ void async_ld16(unsigned short* lds,
                                           const unsigned short* gbase,
                                           unsigned int byteoff) {
    asm volatile("global_load_async_to_lds_b128 %0, %1, %2 offset:0"
                 :
                 : "v"((unsigned int)(uintptr_t)lds),
                   "v"(byteoff),
                   "s"((unsigned long long)(uintptr_t)gbase)
                 : "memory");
}

__device__ __forceinline__ void wait_asynccnt_le4() {
    asm volatile("s_wait_asynccnt 0x4" ::: "memory");
}
__device__ __forceinline__ void wait_asynccnt_0() {
    asm volatile("s_wait_asynccnt 0x0" ::: "memory");
}

// ---------------------------------------------------------------------------
// fp32 -> bf16 elementwise convert (weights; done once per launch so the
// 138 MB bf16 weight set becomes L2-resident for the GEMMs)
// ---------------------------------------------------------------------------
__global__ __launch_bounds__(256) void k_f32_to_bf16(
    const float* __restrict__ in, unsigned short* __restrict__ out, size_t n) {
    size_t i = (size_t)blockIdx.x * blockDim.x + threadIdx.x;
    size_t stride = (size_t)gridDim.x * blockDim.x;
    for (; i < n; i += stride) out[i] = f2bf(in[i]);
}

// ---------------------------------------------------------------------------
// Fused RMSNorm + int8-grid activation quant (quant_input) -> bf16 output.
// One block (256 thr) per row of length n (n = 2560 here, VPT = 10).
// ---------------------------------------------------------------------------
template <int VPT>
__global__ __launch_bounds__(256) void k_rmsnorm_quant(
    const float* __restrict__ X, const float* __restrict__ W,
    unsigned short* __restrict__ Out, int n) {
    __shared__ float red[8];
    const int row = blockIdx.x, tid = threadIdx.x;
    const int lane = tid & 31, wv = tid >> 5;
    const float* xr = X + (size_t)row * n;

    float v[VPT];
    float ss = 0.f;
#pragma unroll
    for (int i = 0; i < VPT; ++i) {
        int c = i * 256 + tid;
        float t = (c < n) ? xr[c] : 0.f;
        v[i] = t; ss += t * t;
    }
#pragma unroll
    for (int o = 16; o > 0; o >>= 1) ss += __shfl_xor(ss, o, 32);
    if (lane == 0) red[wv] = ss;
    __syncthreads();
    float tot = 0.f;
#pragma unroll
    for (int i = 0; i < 8; ++i) tot += red[i];
    const float rs = rsqrtf(tot / (float)n + 1e-5f);
    __syncthreads();

    float amax = 0.f;
#pragma unroll
    for (int i = 0; i < VPT; ++i) {
        int c = i * 256 + tid;
        float h = (c < n) ? v[i] * rs * W[c] : 0.f;
        v[i] = h;
        amax = fmaxf(amax, fabsf(h));
    }
#pragma unroll
    for (int o = 16; o > 0; o >>= 1) amax = fmaxf(amax, __shfl_xor(amax, o, 32));
    if (lane == 0) red[wv] = amax;
    __syncthreads();
    float gmax = 0.f;
#pragma unroll
    for (int i = 0; i < 8; ++i) gmax = fmaxf(gmax, red[i]);
    const float s = 127.f / fmaxf(gmax, 1e-5f);
    const float invs = 1.f / s;
#pragma unroll
    for (int i = 0; i < VPT; ++i) {
        int c = i * 256 + tid;
        if (c < n) {
            float q = rintf(v[i] * s);
            q = fminf(fmaxf(q, -128.f), 127.f);
            Out[(size_t)row * n + c] = f2bf(q * invs);
        }
    }
}

// ---------------------------------------------------------------------------
// Fused FFN inner: g = relu(x1)^2 * x3 ; rmsnorm(g, w) ; quant -> bf16
// Row length FFN = 6912, VPT = 27.
// ---------------------------------------------------------------------------
template <int VPT>
__global__ __launch_bounds__(256) void k_ffn_act_quant(
    const float* __restrict__ X13, const float* __restrict__ W,
    unsigned short* __restrict__ Out, int ffn) {
    __shared__ float red[8];
    const int row = blockIdx.x, tid = threadIdx.x;
    const int lane = tid & 31, wv = tid >> 5;
    const float* xr = X13 + (size_t)row * (2 * ffn);

    float v[VPT];
    float ss = 0.f;
#pragma unroll
    for (int i = 0; i < VPT; ++i) {
        int c = i * 256 + tid;
        float g = 0.f;
        if (c < ffn) {
            float x1 = xr[c], x3 = xr[ffn + c];
            float r = fmaxf(x1, 0.f);
            g = r * r * x3;
        }
        v[i] = g; ss += g * g;
    }
#pragma unroll
    for (int o = 16; o > 0; o >>= 1) ss += __shfl_xor(ss, o, 32);
    if (lane == 0) red[wv] = ss;
    __syncthreads();
    float tot = 0.f;
#pragma unroll
    for (int i = 0; i < 8; ++i) tot += red[i];
    const float rs = rsqrtf(tot / (float)ffn + 1e-5f);
    __syncthreads();

    float amax = 0.f;
#pragma unroll
    for (int i = 0; i < VPT; ++i) {
        int c = i * 256 + tid;
        float h = (c < ffn) ? v[i] * rs * W[c] : 0.f;
        v[i] = h;
        amax = fmaxf(amax, fabsf(h));
    }
#pragma unroll
    for (int o = 16; o > 0; o >>= 1) amax = fmaxf(amax, __shfl_xor(amax, o, 32));
    if (lane == 0) red[wv] = amax;
    __syncthreads();
    float gmax = 0.f;
#pragma unroll
    for (int i = 0; i < 8; ++i) gmax = fmaxf(gmax, red[i]);
    const float s = 127.f / fmaxf(gmax, 1e-5f);
    const float invs = 1.f / s;
#pragma unroll
    for (int i = 0; i < VPT; ++i) {
        int c = i * 256 + tid;
        if (c < ffn) {
            float q = rintf(v[i] * s);
            q = fminf(fmaxf(q, -128.f), 127.f);
            Out[(size_t)row * ffn + c] = f2bf(q * invs);
        }
    }
}

// ---------------------------------------------------------------------------
// bf16 WMMA GEMM:  C[M,N] = A[M,K] * B[N,K]^T  (+ optional residual)
// Block = 256 threads (8 waves), tile 128x128, BK = 32.
// Double-buffered LDS staging via GLOBAL_LOAD_ASYNC_TO_LDS_B128 (ASYNCcnt):
// tile k+1 streams into the alternate buffer while tile k is computed.
// Each thread issues 4 async b128 per tile; s_wait_asynccnt 4 after issuing
// the next tile guarantees (in-order async completion) the current tile is
// resident while the next one is still in flight.
// ---------------------------------------------------------------------------
__device__ __forceinline__ void issue_tile_async(
    unsigned short* bufA, unsigned short* bufB,
    const unsigned short* A, const unsigned short* B,
    int m0, int n0, int K, int k0, int tid) {
#pragma unroll
    for (int j = 0; j < 2; ++j) {
        int c = tid + 256 * j;
        int r = c >> 2, cc = c & 3;
        async_ld16(&bufA[r * 32 + cc * 8], A,
                   (unsigned int)(((size_t)(m0 + r) * K + k0 + cc * 8) * 2));
        async_ld16(&bufB[r * 32 + cc * 8], B,
                   (unsigned int)(((size_t)(n0 + r) * K + k0 + cc * 8) * 2));
    }
}

__global__ __launch_bounds__(256) void k_gemm_bf16(
    const unsigned short* __restrict__ A, const unsigned short* __restrict__ B,
    const float* __restrict__ Res, float* __restrict__ C,
    int M, int N, int K) {
    __shared__ unsigned short sA[2][128 * 32];   // 2 x 8 KB
    __shared__ unsigned short sB[2][128 * 32];   // 2 x 8 KB

    const int tid  = threadIdx.x;
    const int lane = tid & 31;
    const int wv   = tid >> 5;
    const int wM   = wv & 3;       // 0..3
    const int wN   = wv >> 2;      // 0..1
    const int l16  = lane & 15;
    const int hlf  = lane >> 4;    // 0/1

    const int m0 = blockIdx.y * 128;
    const int n0 = blockIdx.x * 128;

    floatx8 acc[2][4] = {};

    const int nk = K >> 5;
    // prologue: start tile 0
    issue_tile_async(sA[0], sB[0], A, B, m0, n0, K, 0, tid);

    for (int t = 0; t < nk; ++t) {
        const int cur = t & 1;
        __syncthreads();   // alt buffer free (tile t-1 consumed by everyone)
        if (t + 1 < nk) {
            issue_tile_async(sA[cur ^ 1], sB[cur ^ 1], A, B, m0, n0, K,
                             (t + 1) << 5, tid);
            wait_asynccnt_le4();   // tile t landed; tile t+1 still in flight
        } else {
            wait_asynccnt_0();
        }
        __syncthreads();   // everyone's tile-t DMA complete

        const unsigned short* cA = sA[cur];
        const unsigned short* cB = sB[cur];

        bf16x16 aF[2];
#pragma unroll
        for (int mi = 0; mi < 2; ++mi) {
            int base = (wM * 32 + mi * 16 + l16) * 32 + hlf * 16;
            bf16x8 lo = *(const bf16x8*)&cA[base];
            bf16x8 hi = *(const bf16x8*)&cA[base + 8];
            aF[mi] = pack16(lo, hi);
        }
#pragma unroll
        for (int ni = 0; ni < 4; ++ni) {
            int nb = (wN * 64 + ni * 16 + l16) * 32 + hlf * 16;
            bf16x16 bF = *(const bf16x16*)&cB[nb];
#pragma unroll
            for (int mi = 0; mi < 2; ++mi) {
                acc[mi][ni] = __builtin_amdgcn_wmma_f32_16x16x32_bf16(
                    false, aF[mi], false, bF, (short)0, acc[mi][ni], false, false);
            }
        }
    }

    // epilogue: C layout (lane L: col = L%16, rows r + 8*(L>=16))
#pragma unroll
    for (int mi = 0; mi < 2; ++mi) {
#pragma unroll
        for (int ni = 0; ni < 4; ++ni) {
#pragma unroll
            for (int r = 0; r < 8; ++r) {
                int row = m0 + wM * 32 + mi * 16 + r + 8 * hlf;
                int col = n0 + wN * 64 + ni * 16 + l16;
                size_t idx = (size_t)row * N + col;
                float val = acc[mi][ni][r];
                if (Res) val += Res[idx];
                C[idx] = val;
            }
        }
    }
}

// ---------------------------------------------------------------------------
// RoPE + bf16 repack of Q/K/V out of the fused QKV GEMM output.
// QKV layout: [30][T][128] bf16 : heads 0..19 = Q(rope), 20..24 = K(rope),
//                                 25..29 = V(copy).
// ---------------------------------------------------------------------------
__global__ __launch_bounds__(256) void k_rope_split(
    const float* __restrict__ XQKV, unsigned short* __restrict__ QKV) {
    const size_t total = (size_t)T_SEQ * 30 * 64;
    size_t idx = (size_t)blockIdx.x * 256 + threadIdx.x;
    if (idx >= total) return;
    int p = (int)(idx & 63);
    int h = (int)((idx >> 6) % 30);
    int t = (int)(idx / (64 * 30));

    const float* src = XQKV + (size_t)t * QKV_D + h * HEAD_D + 2 * p;
    float x1 = src[0], x2 = src[1];
    unsigned short o1, o2;
    if (h < 25) {  // rope for q and k
        float inv = __powf(500000.0f, -(float)p * (1.0f / 64.0f));
        float sn, cs;
        __sincosf((float)t * inv, &sn, &cs);
        o1 = f2bf(x1 * cs - x2 * sn);
        o2 = f2bf(x1 * sn + x2 * cs);
    } else {       // v: plain convert
        o1 = f2bf(x1);
        o2 = f2bf(x2);
    }
    unsigned short* dst = QKV + ((size_t)h * T_SEQ + t) * HEAD_D + 2 * p;
    dst[0] = o1; dst[1] = o2;
}

// ---------------------------------------------------------------------------
// Flash attention (causal, GQA: kv = head>>2), all matmuls via WMMA bf16.
// Grid: (T/128, N_HEADS). Block 256 = 8 waves; each wave owns 16 query rows.
// ---------------------------------------------------------------------------
__global__ __launch_bounds__(256) void k_flash_attn(
    const unsigned short* __restrict__ QKV, float* __restrict__ Attn) {
    __shared__ unsigned short sVT[128 * 128];      // 32 KB : [d][krow]
    __shared__ unsigned short sP[8 * 16 * 128];    // 32 KB : per-wave 16x128

    const int qblk = blockIdx.x;
    const int head = blockIdx.y;
    const int kv   = head >> 2;

    const int tid  = threadIdx.x;
    const int lane = tid & 31;
    const int wv   = tid >> 5;
    const int l16  = lane & 15;
    const int hlf  = lane >> 4;

    const unsigned short* Qb = QKV;
    const unsigned short* Kb = QKV + (size_t)N_HEADS * T_SEQ * HEAD_D;
    const unsigned short* Vb = Kb + (size_t)N_KV * T_SEQ * HEAD_D;

    const int qrow0 = qblk * 128 + wv * 16;

    // Q fragments (rows qrow0..+15, full head_dim in 4 K-steps) held in regs
    bf16x16 aQ[4];
    {
        const unsigned short* qr = Qb + ((size_t)head * T_SEQ + qrow0 + l16) * HEAD_D;
#pragma unroll
        for (int ks = 0; ks < 4; ++ks) {
            int d0 = ks * 32 + hlf * 16;
            bf16x8 lo = *(const bf16x8*)(qr + d0);
            bf16x8 hi = *(const bf16x8*)(qr + d0 + 8);
            aQ[ks] = pack16(lo, hi);
        }
    }

    float m_run[8], l_run[8];
#pragma unroll
    for (int r = 0; r < 8; ++r) { m_run[r] = -1e30f; l_run[r] = 0.f; }
    floatx8 o[8] = {};

    const float scale = 0.08838834764831845f;  // 1/sqrt(128)
    unsigned short* myP = &sP[wv * (16 * 128)];

    for (int kb = 0; kb <= qblk; ++kb) {
        // ---- stage V^T chunk: sVT[d][krow] ----
        __syncthreads();
#pragma unroll
        for (int j = 0; j < 8; ++j) {
            int id = tid + 256 * j;
            int vr = id >> 4, cc = id & 15;
            uint4 dv = *(const uint4*)(Vb + ((size_t)kv * T_SEQ + kb * 128 + vr) * HEAD_D + cc * 8);
            const unsigned short* pv = (const unsigned short*)&dv;
#pragma unroll
            for (int i = 0; i < 8; ++i) sVT[(cc * 8 + i) * 128 + vr] = pv[i];
        }
        __syncthreads();

        // ---- S = Q K^T (scaled) ----
        floatx8 sAcc[8] = {};
#pragma unroll
        for (int ni = 0; ni < 8; ++ni) {
            const unsigned short* kr =
                Kb + ((size_t)kv * T_SEQ + kb * 128 + ni * 16 + l16) * HEAD_D;
#pragma unroll
            for (int ks = 0; ks < 4; ++ks) {
                bf16x16 bF = *(const bf16x16*)(kr + ks * 32 + hlf * 16);
                sAcc[ni] = __builtin_amdgcn_wmma_f32_16x16x32_bf16(
                    false, aQ[ks], false, bF, (short)0, sAcc[ni], false, false);
            }
        }
        // scale + causal mask (only the diagonal chunk needs masking)
        const bool diag = (kb == qblk);
#pragma unroll
        for (int ni = 0; ni < 8; ++ni) {
#pragma unroll
            for (int r = 0; r < 8; ++r) {
                float s = sAcc[ni][r] * scale;
                if (diag) {
                    int col = kb * 128 + ni * 16 + l16;
                    int row = qrow0 + r + 8 * hlf;
                    if (col > row) s = -1e30f;
                }
                sAcc[ni][r] = s;
            }
        }

        // ---- online softmax (rows live in one 16-lane half; shfl 1/2/4/8) ----
        float f[8], rsum[8], mnew[8];
#pragma unroll
        for (int r = 0; r < 8; ++r) {
            float mx = -1e30f;
#pragma unroll
            for (int ni = 0; ni < 8; ++ni) mx = fmaxf(mx, sAcc[ni][r]);
#pragma unroll
            for (int off = 1; off < 16; off <<= 1)
                mx = fmaxf(mx, __shfl_xor(mx, off, 32));
            mnew[r] = fmaxf(m_run[r], mx);
            f[r] = __expf(m_run[r] - mnew[r]);
            m_run[r] = mnew[r];
            rsum[r] = 0.f;
        }
#pragma unroll
        for (int ni = 0; ni < 8; ++ni) {
#pragma unroll
            for (int r = 0; r < 8; ++r) {
                float p = __expf(sAcc[ni][r] - mnew[r]);
                sAcc[ni][r] = p;
                rsum[r] += p;
            }
        }
#pragma unroll
        for (int r = 0; r < 8; ++r) {
#pragma unroll
            for (int off = 1; off < 16; off <<= 1)
                rsum[r] += __shfl_xor(rsum[r], off, 32);
            l_run[r] = l_run[r] * f[r] + rsum[r];
        }
        // rescale O
#pragma unroll
        for (int ni = 0; ni < 8; ++ni)
#pragma unroll
            for (int r = 0; r < 8; ++r) o[ni][r] *= f[r];

        // ---- P -> wave-private LDS (C-layout -> row-major) ----
#pragma unroll
        for (int ni = 0; ni < 8; ++ni)
#pragma unroll
            for (int r = 0; r < 8; ++r)
                myP[(r + 8 * hlf) * 128 + ni * 16 + l16] = f2bf(sAcc[ni][r]);
        // wave-private region: LDS is in-order within a wave, no barrier needed

        // ---- O += P V ----
#pragma unroll
        for (int ks = 0; ks < 4; ++ks) {
            const unsigned short* pr = myP + l16 * 128 + ks * 32 + hlf * 16;
            bf16x8 lo = *(const bf16x8*)pr;
            bf16x8 hi = *(const bf16x8*)(pr + 8);
            bf16x16 aP = pack16(lo, hi);
#pragma unroll
            for (int ni = 0; ni < 8; ++ni) {
                bf16x16 bV = *(const bf16x16*)&sVT[(ni * 16 + l16) * 128 + ks * 32 + hlf * 16];
                o[ni] = __builtin_amdgcn_wmma_f32_16x16x32_bf16(
                    false, aP, false, bV, (short)0, o[ni], false, false);
            }
        }
    }

    // ---- finalize: O / l, write fp32 [T, 2560] with head-major columns ----
#pragma unroll
    for (int r = 0; r < 8; ++r) {
        float inv_l = 1.f / l_run[r];
        int row = qrow0 + r + 8 * hlf;
#pragma unroll
        for (int ni = 0; ni < 8; ++ni) {
            int col = head * HEAD_D + ni * 16 + l16;
            Attn[(size_t)row * DIM + col] = o[ni][r] * inv_l;
        }
    }
}

// ---------------------------------------------------------------------------
// Launcher
// ---------------------------------------------------------------------------
extern "C" void kernel_launch(void* const* d_in, const int* in_sizes, int n_in,
                              void* d_out, int out_size, void* d_ws, size_t ws_size,
                              hipStream_t stream) {
    (void)in_sizes; (void)n_in; (void)out_size; (void)ws_size;
    const float* x    = (const float*)d_in[0];
    const float* wqkv = (const float*)d_in[1];
    const float* wo   = (const float*)d_in[2];
    const float* w13  = (const float*)d_in[3];
    const float* w2   = (const float*)d_in[4];
    const float* anw  = (const float*)d_in[5];
    const float* fnw  = (const float*)d_in[6];
    const float* asnw = (const float*)d_in[7];
    const float* fsnw = (const float*)d_in[8];
    float* out = (float*)d_out;

    char* ws = (char*)d_ws;
    auto bump = [&](size_t bytes) -> void* {
        void* p = (void*)ws;
        ws += (bytes + 255) & ~(size_t)255;
        return p;
    };
    unsigned short* wqkv_b = (unsigned short*)bump((size_t)QKV_D * DIM * 2);
    unsigned short* wo_b   = (unsigned short*)bump((size_t)DIM * DIM * 2);
    unsigned short* w13_b  = (unsigned short*)bump((size_t)2 * FFN * DIM * 2);
    unsigned short* w2_b   = (unsigned short*)bump((size_t)DIM * FFN * 2);
    unsigned short* act_b  = (unsigned short*)bump((size_t)T_SEQ * DIM * 2);
    float*          xqkv   = (float*)bump((size_t)T_SEQ * QKV_D * 4);
    unsigned short* qkv_b  = (unsigned short*)bump((size_t)30 * T_SEQ * HEAD_D * 2);
    float*          attn   = (float*)bump((size_t)T_SEQ * DIM * 4);
    float*          hbuf   = (float*)bump((size_t)T_SEQ * DIM * 4);
    float*          x13    = (float*)bump((size_t)T_SEQ * 2 * FFN * 4);
    unsigned short* inner_b= (unsigned short*)bump((size_t)T_SEQ * FFN * 2);

    // 0) one-time (per launch) weight conversion to bf16 -> L2-resident
    k_f32_to_bf16<<<4096, 256, 0, stream>>>(wqkv, wqkv_b, (size_t)QKV_D * DIM);
    k_f32_to_bf16<<<4096, 256, 0, stream>>>(wo,   wo_b,   (size_t)DIM * DIM);
    k_f32_to_bf16<<<8192, 256, 0, stream>>>(w13,  w13_b,  (size_t)2 * FFN * DIM);
    k_f32_to_bf16<<<4096, 256, 0, stream>>>(w2,   w2_b,   (size_t)DIM * FFN);

    // 1) h = quant(rmsnorm(x, attn_norm_w)) -> bf16
    k_rmsnorm_quant<10><<<T_SEQ, 256, 0, stream>>>(x, anw, act_b, DIM);

    // 2) xqkv = h @ wqkv^T
    k_gemm_bf16<<<dim3(QKV_D / 128, T_SEQ / 128), 256, 0, stream>>>(
        act_b, wqkv_b, nullptr, xqkv, T_SEQ, QKV_D, DIM);

    // 3) RoPE + bf16 split into Q/K/V
    {
        size_t total = (size_t)T_SEQ * 30 * 64;
        k_rope_split<<<(unsigned)((total + 255) / 256), 256, 0, stream>>>(xqkv, qkv_b);
    }

    // 4) causal GQA flash attention -> attn fp32 [T, DIM]
    k_flash_attn<<<dim3(T_SEQ / 128, N_HEADS), 256, 0, stream>>>(qkv_b, attn);

    // 5) attn_q = quant(rmsnorm(attn, attn_sub_norm_w))
    k_rmsnorm_quant<10><<<T_SEQ, 256, 0, stream>>>(attn, asnw, act_b, DIM);

    // 6) h = x + attn_q @ wo^T
    k_gemm_bf16<<<dim3(DIM / 128, T_SEQ / 128), 256, 0, stream>>>(
        act_b, wo_b, x, hbuf, T_SEQ, DIM, DIM);

    // 7) hn = quant(rmsnorm(h, ffn_norm_w))
    k_rmsnorm_quant<10><<<T_SEQ, 256, 0, stream>>>(hbuf, fnw, act_b, DIM);

    // 8) x13 = hn @ w13^T
    k_gemm_bf16<<<dim3((2 * FFN) / 128, T_SEQ / 128), 256, 0, stream>>>(
        act_b, w13_b, nullptr, x13, T_SEQ, 2 * FFN, DIM);

    // 9) inner = quant(rmsnorm(relu(x1)^2 * x3, ffn_sub_norm_w))
    k_ffn_act_quant<27><<<T_SEQ, 256, 0, stream>>>(x13, fsnw, inner_b, FFN);

    // 10) out = h + inner @ w2^T
    k_gemm_bf16<<<dim3(DIM / 128, T_SEQ / 128), 256, 0, stream>>>(
        inner_b, w2_b, hbuf, out, T_SEQ, DIM, FFN);
}
